// MultiHeadAttention_84052509983487
// MI455X (gfx1250) — matrix operand order repaired and checked by reading, not compiled, not run on hardware
//
#include <hip/hip_runtime.h>
#include <hip/hip_bf16.h>

// MHA forward for B=4, S=2048, H=512, NH=8, HD=64 on gfx1250 (wave32, WMMA).
// d_out = [hidden (B*S*H f32)] ++ [attn_probs (B*NH*S*S f32)]
// d_ws  = Qh, Kh (f16 [B,S,H]), Vt (f16 [B,NH,HD,S]), Ch (f16 [B,S,H]) = 32 MB

#define BB 4
#define SS 2048
#define HH 512
#define NHH 8
#define HDD 64

typedef __attribute__((ext_vector_type(16))) _Float16 v16h;
typedef __attribute__((ext_vector_type(8)))  _Float16 v8h;
typedef __attribute__((ext_vector_type(8)))  float    v8f;

static __device__ __forceinline__ v8f wmma_f16(v16h a, v16h b, v8f c) {
  return __builtin_amdgcn_wmma_f32_16x16x32_f16(false, a, false, b, (short)0, c,
                                                false, false);
}

// A fragment (16x32, f16) from row-major f32 source; p -> (row0, k0), ld elems.
// ISA layout: elems 0..7 = K[lh*8 + j], elems 8..15 = K[16 + lh*8 + j], row = lane&15.
static __device__ __forceinline__ v16h load_a_f32(const float* p, int ld, int lane) {
  const int lr = lane & 15, lh = lane >> 4;
  const float* q = p + (size_t)lr * ld + lh * 8;
  v16h a;
#pragma unroll
  for (int j = 0; j < 8; ++j) {
    a[j]     = (_Float16)q[j];
    a[j + 8] = (_Float16)q[j + 16];
  }
  return a;
}

static __device__ __forceinline__ v16h load_a_f16(const _Float16* p, int ld, int lane) {
  const int lr = lane & 15, lh = lane >> 4;
  const _Float16* q = p + (size_t)lr * ld + lh * 8;
  v8h lo = *(const v8h*)q;
  v8h hi = *(const v8h*)(q + 16);
  v16h a;
#pragma unroll
  for (int j = 0; j < 8; ++j) { a[j] = lo[j]; a[j + 8] = hi[j]; }
  return a;
}

// B fragment (32x16, f16): column n = lane&15, elems 0..7 = K[lh*8 + j], elems 8..15 = K[16 + lh*8 + j].
// Source "row-major by N" (weight rows / key rows / Vt rows): p -> (n0, k0).
static __device__ __forceinline__ v16h load_b_f32(const float* p, int ld, int lane) {
  const int lr = lane & 15, lh = lane >> 4;
  const float* q = p + (size_t)lr * ld + lh * 8;
  v16h b;
#pragma unroll
  for (int j = 0; j < 8; ++j) { b[j] = (_Float16)q[j]; b[j + 8] = (_Float16)q[j + 16]; }
  return b;
}

static __device__ __forceinline__ v16h load_b_f16(const _Float16* p, int ld, int lane) {
  const int lr = lane & 15, lh = lane >> 4;
  const _Float16* q = p + (size_t)lr * ld + lh * 8;
  v8h lo = *(const v8h*)q;
  v8h hi = *(const v8h*)(q + 16);
  v16h b;
#pragma unroll
  for (int j = 0; j < 8; ++j) { b[j] = lo[j]; b[j + 8] = hi[j]; }
  return b;
}

// ---------------- QKV projection: out = X @ W^T + bias, stored f16 ----------
__global__ __launch_bounds__(128) void qkv_proj_kernel(
    const float* __restrict__ query, const float* __restrict__ key,
    const float* __restrict__ value, const float* __restrict__ Wq,
    const float* __restrict__ bq, const float* __restrict__ Wv,
    const float* __restrict__ bv, _Float16* __restrict__ Qh,
    _Float16* __restrict__ Kh, _Float16* __restrict__ Vt) {
  const int lane = threadIdx.x & 31;
  const int gw = (blockIdx.x * blockDim.x + threadIdx.x) >> 5;
  const int NT = HH / 16;                    // 32
  const int TPM = (BB * SS / 16) * NT;       // 16384 tiles per matrix
  const int mat = gw / TPM;                  // 0=Q 1=K(uses Wq: source bug) 2=V
  const int t = gw % TPM;
  const int m0 = (t / NT) * 16, n0 = (t % NT) * 16;
  const float* X = (mat == 0) ? query : (mat == 1) ? key : value;
  const float* W = (mat == 2) ? Wv : Wq;
  const float* bias = (mat == 2) ? bv : bq;
  const int lr = lane & 15, lh = lane >> 4;

  v8f c = {};
  for (int k0 = 0; k0 < HH; k0 += 32) {
    v16h a = load_a_f32(X + (size_t)m0 * HH + k0, HH, lane);
    v16h b = load_b_f32(W + (size_t)n0 * HH + k0, HH, lane);
    c = wmma_f16(a, b, c);
  }
  const float bval = bias[n0 + lr];
#pragma unroll
  for (int r = 0; r < 8; ++r) {
    const int m = m0 + r + 8 * lh;
    const int n = n0 + lr;
    const float o = c[r] + bval;
    if (mat == 0) {
      Qh[(size_t)m * HH + n] = (_Float16)o;
    } else if (mat == 1) {
      Kh[(size_t)m * HH + n] = (_Float16)o;
    } else {  // V stored transposed per head: Vt[b][h][d][s]
      const int bi = m / SS, s = m % SS;
      const int h = n / HDD, d = n % HDD;
      Vt[(((size_t)bi * NHH + h) * HDD + d) * SS + s] = (_Float16)o;
    }
  }
}

// ---------------- Attention: scores + mask + softmax + probs out + ctx ------
#define SC_LD 2052  // +4 floats pad: conflict-free 16-lane x b128 column tiles

__global__ __launch_bounds__(128) void attn_kernel(
    const _Float16* __restrict__ Qh, const _Float16* __restrict__ Kh,
    const _Float16* __restrict__ Vt, const int* __restrict__ pm,
    float* __restrict__ probs, _Float16* __restrict__ Ch) {
  __shared__ float sc[16][SC_LD];   // unnormalized exp(scores) strip
  __shared__ float redm[16][8];
  __shared__ float reds[16][8];
  __shared__ float invs[16];

  const int lane = threadIdx.x & 31;
  const int wid = threadIdx.x >> 5;
  const int lr = lane & 15, lh = lane >> 4;
  const int qt = blockIdx.x & 127;   // S/16 tiles
  const int bh = blockIdx.x >> 7;
  const int b = bh / NHH, h = bh % NHH;
  const int m0 = qt * 16;

  // Q fragments for head-dim chunks kc=0 (d 0..31), kc=1 (d 32..63)
  const _Float16* qbase = Qh + ((size_t)b * SS + m0) * HH + h * HDD;
  const v16h a0 = load_a_f16(qbase + 0, HH, lane);
  const v16h a1 = load_a_f16(qbase + 32, HH, lane);

  const float scale = 0.125f;  // 1/sqrt(64)
  const int* pmb = pm + ((size_t)b * SS + m0) * SS;  // pad_mask (int per doc)
  const _Float16* kbase = Kh + (size_t)b * SS * HH + h * HDD;

  // ---- scores: each wave strides over key tiles
  for (int i = wid; i < SS / 16; i += 4) {
    const int n0 = i * 16;
    // prefetch next tile for this wave (global_prefetch_b8)
    if (i + 4 < SS / 16) {
      __builtin_prefetch(kbase + (size_t)(n0 + 64 + lr) * HH, 0, 1);
    }
    v16h b0 = load_b_f16(kbase + (size_t)n0 * HH + 0, HH, lane);
    v16h b1 = load_b_f16(kbase + (size_t)n0 * HH + 32, HH, lane);
    v8f c = {};
    c = wmma_f16(a0, b0, c);
    c = wmma_f16(a1, b1, c);
    const int n = n0 + lr;
#pragma unroll
    for (int r = 0; r < 8; ++r) {
      const int row = r + 8 * lh;
      float v = c[r] * scale;
      // mask is streamed once: non-temporal load
      if (__builtin_nontemporal_load(&pmb[(size_t)row * SS + n]) != 0) v = -1e9f;
      sc[row][n] = v;
    }
  }
  __syncthreads();

  // ---- softmax: 8 threads per row, 256 elems each (staggered start per seg)
  const int row = threadIdx.x >> 3;
  const int seg = threadIdx.x & 7;
  float mx = -3.0e38f;
  for (int j = 0; j < 256; ++j)
    mx = fmaxf(mx, sc[row][seg * 256 + ((j + seg * 32) & 255)]);
  redm[row][seg] = mx;
  __syncthreads();
  float rm = redm[row][0];
#pragma unroll
  for (int k = 1; k < 8; ++k) rm = fmaxf(rm, redm[row][k]);
  float sum = 0.f;
  for (int j = 0; j < 256; ++j) {
    const int idx = seg * 256 + ((j + seg * 32) & 255);
    const float e = __expf(sc[row][idx] - rm);
    sc[row][idx] = e;
    sum += e;
  }
  reds[row][seg] = sum;
  __syncthreads();
  if (threadIdx.x < 16) {
    float s = 0.f;
#pragma unroll
    for (int k = 0; k < 8; ++k) s += reds[threadIdx.x][k];
    invs[threadIdx.x] = 1.0f / s;
  }
  __syncthreads();

  // ---- write normalized probs, coalesced, non-temporal (512 MB stream,
  //      larger than the 192 MB L2 -- keep it out of cache)
  float* pout = probs + (((size_t)b * NHH + h) * SS + m0) * SS;
  for (int e = threadIdx.x; e < 16 * SS; e += 128) {
    const int rw = e >> 11, cl = e & (SS - 1);
    __builtin_nontemporal_store(sc[rw][cl] * invs[rw], &pout[(size_t)rw * SS + cl]);
  }

  // ---- ctx = probs @ V : each wave owns a 16-wide head-dim slice
  const int dc0 = wid * 16;
  const _Float16* vbase = Vt + (((size_t)b * NHH + h) * HDD + dc0) * SS;
  const float inv = invs[lr];  // A-fragment row for this lane
  v8f acc = {};
  for (int k0 = 0; k0 < SS; k0 += 32) {
    v16h a;
    const float* p1 = &sc[lr][k0 + lh * 8];
    const float* p2 = &sc[lr][k0 + 16 + lh * 8];
#pragma unroll
    for (int j = 0; j < 8; ++j) {
      a[j]     = (_Float16)(p1[j] * inv);
      a[j + 8] = (_Float16)(p2[j] * inv);
    }
    v16h bm = load_b_f16(vbase + k0, SS, lane);  // contiguous thanks to Vt
    acc = wmma_f16(a, bm, acc);
  }
#pragma unroll
  for (int r = 0; r < 8; ++r) {
    const int m = m0 + r + 8 * lh;
    Ch[((size_t)b * SS + m) * HH + h * HDD + dc0 + lr] = (_Float16)acc[r];
  }
}

// ---------------- Output projection: hidden = ctx @ Wd^T + bd (f32 out) -----
__global__ __launch_bounds__(128) void out_proj_kernel(
    const _Float16* __restrict__ Ch, const float* __restrict__ Wd,
    const float* __restrict__ bd, float* __restrict__ out) {
  const int lane = threadIdx.x & 31;
  const int gw = (blockIdx.x * blockDim.x + threadIdx.x) >> 5;
  const int NT = HH / 16;
  const int m0 = (gw / NT) * 16, n0 = (gw % NT) * 16;
  const int lr = lane & 15, lh = lane >> 4;
  v8f c = {};
  for (int k0 = 0; k0 < HH; k0 += 32) {
    v16h a = load_a_f16(Ch + (size_t)m0 * HH + k0, HH, lane);
    v16h b = load_b_f32(Wd + (size_t)n0 * HH + k0, HH, lane);
    c = wmma_f16(a, b, c);
  }
  const float bval = bd[n0 + lr];
#pragma unroll
  for (int r = 0; r < 8; ++r)
    out[((size_t)m0 + r + 8 * lh) * HH + n0 + lr] = c[r] + bval;
}

extern "C" void kernel_launch(void* const* d_in, const int* in_sizes, int n_in,
                              void* d_out, int out_size, void* d_ws, size_t ws_size,
                              hipStream_t stream) {
  const float* query = (const float*)d_in[0];
  const float* key   = (const float*)d_in[1];
  const float* value = (const float*)d_in[2];
  const int*   pad   = (const int*)d_in[3];
  const float* Wq = (const float*)d_in[4];
  const float* bq = (const float*)d_in[5];
  const float* Wv = (const float*)d_in[6];
  const float* bv = (const float*)d_in[7];
  const float* Wd = (const float*)d_in[8];
  const float* bd = (const float*)d_in[9];

  float* out   = (float*)d_out;
  float* probs = out + (size_t)BB * SS * HH;

  _Float16* Qh = (_Float16*)d_ws;
  _Float16* Kh = Qh + (size_t)BB * SS * HH;
  _Float16* Vt = Kh + (size_t)BB * SS * HH;
  _Float16* Ch = Vt + (size_t)BB * SS * HH;

  {
    const int waves = 3 * (BB * SS / 16) * (HH / 16);  // 49152 tiles
    qkv_proj_kernel<<<dim3(waves / 4), dim3(128), 0, stream>>>(
        query, key, value, Wq, bq, Wv, bv, Qh, Kh, Vt);
  }
  {
    attn_kernel<<<dim3(BB * NHH * (SS / 16)), dim3(128), 0, stream>>>(
        Qh, Kh, Vt, pad, probs, Ch);
  }
  {
    const int waves = (BB * SS / 16) * (HH / 16);  // 16384 tiles
    out_proj_kernel<<<dim3(waves / 4), dim3(128), 0, stream>>>(Ch, Wd, bd, out);
  }
}
